// GINModelWithResidual_31628139167863
// MI455X (gfx1250) — matrix operand order repaired and checked by reading, not compiled, not run on hardware
//
#include <hip/hip_runtime.h>

#define D 128
#define THREADS 256
#define WPITCH 144                       // ushorts per LDS row: 128 + 16 (32B pad)
#define SMEM_BYTES (2 * 128 * WPITCH * 2)  // hi + lo staged W^T, 73728 B

typedef __attribute__((ext_vector_type(16))) __bf16 v16bf;
typedef __attribute__((ext_vector_type(8)))  float  v8f;
typedef __attribute__((ext_vector_type(4)))  unsigned int u32x4;
typedef __attribute__((ext_vector_type(8)))  int i32x8;
typedef __attribute__((ext_vector_type(4)))  int i32x4;

union Frag {
    v16bf v;
    uint4 q[2];
};

__device__ __forceinline__ unsigned short f2bf(float x) {
    unsigned int u = __float_as_uint(x);
    u += 0x7FFFu + ((u >> 16) & 1u);            // round-to-nearest-even
    return (unsigned short)(u >> 16);
}
__device__ __forceinline__ float bf2f(unsigned short h) {
    return __uint_as_float(((unsigned int)h) << 16);
}

// Split fp32 activations into bf16 hi + bf16 lo (x ~= hi + lo).
__global__ void split_act_kernel(const float* __restrict__ x,
                                 unsigned short* __restrict__ hi,
                                 unsigned short* __restrict__ lo,
                                 int total) {
    int idx = blockIdx.x * blockDim.x + threadIdx.x;
    if (idx >= total) return;
    float v = x[idx];
    unsigned short h = f2bf(v);
    hi[idx] = h;
    lo[idx] = f2bf(v - bf2f(h));
}

// Split fp32 weight [K x N] into transposed bf16 hi/lo [N x K].
__global__ void split_w_kernel(const float* __restrict__ w,
                               unsigned short* __restrict__ hi_t,
                               unsigned short* __restrict__ lo_t) {
    int idx = blockIdx.x * blockDim.x + threadIdx.x;
    if (idx >= D * D) return;
    int k = idx >> 7;
    int n = idx & (D - 1);
    float v = w[idx];
    unsigned short h = f2bf(v);
    hi_t[n * D + k] = h;
    lo_t[n * D + k] = f2bf(v - bf2f(h));
}

// z[dst] += h[src] : one wave per edge (scalar src/dst), 4 floats per thread.
__global__ void edge_scatter_kernel(const float* __restrict__ h,
                                    const int* __restrict__ src,
                                    const int* __restrict__ dst,
                                    float* __restrict__ z, int nedges) {
    unsigned int tid = blockIdx.x * blockDim.x + threadIdx.x;
    unsigned int work = (unsigned int)nedges * 32u;
    if (tid >= work) return;
    int e  = (int)(tid >> 5);
    int d4 = (int)(tid & 31u) * 4;
    int s = src[e];
    int t = dst[e];
    const float4 v = *(const float4*)(h + (size_t)s * D + d4);
    float* p = z + (size_t)t * D + d4;
    atomicAdd(p + 0, v.x);
    atomicAdd(p + 1, v.y);
    atomicAdd(p + 2, v.z);
    atomicAdd(p + 3, v.w);
}

#if __has_builtin(__builtin_amdgcn_tensor_load_to_lds) && \
    __has_builtin(__builtin_amdgcn_s_wait_tensorcnt)
#define USE_TDM 1
// D# group 0: count=1 | lds_addr | global_addr | type=2 ("image").
__device__ __forceinline__ u32x4 tdm_group0(unsigned int lds_off, const void* gaddr) {
    unsigned long long ga = (unsigned long long)gaddr;
    u32x4 g;
    g.x = 1u;                                              // count=1, user desc
    g.y = lds_off;                                         // lds_addr [63:32]
    g.z = (unsigned int)(ga & 0xFFFFFFFFu);                // global_addr [95:64]
    g.w = (unsigned int)((ga >> 32) & 0x01FFFFFFu) | (2u << 30);  // [120:96] | type
    return g;
}
// D# group 1: 2-D 128x128 tile of 2-byte elems, LDS pad 8 DWORDs every 256 B.
__device__ __forceinline__ i32x8 tdm_group1() {
    i32x8 g;
    g[0] = (1 << 16)      // data_size = 2 bytes
         | (1 << 20)      // pad_enable
         | (5 << 22)      // pad_interval: 64 DWORDs = 256 B (one tile row)
         | (7 << 25);     // pad_amount: 8 DWORDs = 32 B
    g[1] = (int)(128u << 16);   // tensor_dim0[15:0] at bits 63:48
    g[2] = (int)(128u << 16);   // tensor_dim1[15:0] at bits 95:80 (dim0 hi = 0)
    g[3] = (int)(128u << 16);   // tile_dim0 = 128 at bits 127:112
    g[4] = 128;                 // tile_dim1 = 128, tile_dim2 = 0
    g[5] = 128;                 // tensor_dim0_stride low 32 = 128 elements
    g[6] = 0;
    g[7] = 0;
    return g;
}
#else
#define USE_TDM 0
#endif

// out = relu(X @ W + bias) (+ resid). X as bf16 hi/lo, W^T as bf16 hi/lo staged
// into padded LDS (TDM DMA when available). 3-WMMA fp32 emulation, f32 accum.
// Either writes f32 (outF, optional resid) or bf16 hi/lo (outHi/outLo) fused
// with the split for the next GEMM. Wave = 16 rows, block = 8 waves = 128 rows.
__global__ __launch_bounds__(THREADS)
void gemm_kernel(const unsigned short* __restrict__ Ahi,
                 const unsigned short* __restrict__ Alo,
                 const unsigned short* __restrict__ Bhi_t,
                 const unsigned short* __restrict__ Blo_t,
                 const float* __restrict__ bias,
                 const float* __restrict__ resid,      // nullable
                 float* __restrict__ outF,             // nullable
                 unsigned short* __restrict__ outHi,   // nullable
                 unsigned short* __restrict__ outLo,   // nullable
                 int nrows) {
    extern __shared__ unsigned short smem[];
    unsigned short* sWh = smem;                   // [128][WPITCH]
    unsigned short* sWl = smem + 128 * WPITCH;

    const int lane = threadIdx.x & 31;
    const int wave = threadIdx.x >> 5;

#if USE_TDM
    if (wave == 0) {
        const i32x4 z4 = {0, 0, 0, 0};
        const i32x8 z8 = {0, 0, 0, 0, 0, 0, 0, 0};
        const i32x8 g1 = tdm_group1();
        __builtin_amdgcn_tensor_load_to_lds(tdm_group0(0, Bhi_t), g1, z4, z4, z8, 0);
        __builtin_amdgcn_tensor_load_to_lds(tdm_group0(128 * WPITCH * 2, Blo_t),
                                            g1, z4, z4, z8, 0);
        __builtin_amdgcn_s_wait_tensorcnt(0);
    }
    __syncthreads();
#else
    for (int chunk = threadIdx.x; chunk < 128 * 16; chunk += THREADS) {
        int row = chunk >> 4;
        int c   = chunk & 15;
        *(uint4*)(sWh + row * WPITCH + c * 8) = *(const uint4*)(Bhi_t + row * D + c * 8);
        *(uint4*)(sWl + row * WPITCH + c * 8) = *(const uint4*)(Blo_t + row * D + c * 8);
    }
    __syncthreads();
#endif

    const int m0 = blockIdx.x * 128 + wave * 16;

    // A fragment (16x32 bf16): lanes 0-15 hold M=lane, K={kb..kb+7, kb+16..kb+23},
    // kb = (lane>>4)*8.
    const int rowA = m0 + (lane & 15);
    const int rA   = rowA < nrows ? rowA : (nrows - 1);   // clamp; EXEC stays full
    const int kbA  = (lane >> 4) * 8;

    Frag ah[4], al[4];
#pragma unroll
    for (int ks = 0; ks < 4; ++ks) {
        const size_t base = (size_t)rA * D + ks * 32 + kbA;
        ah[ks].q[0] = *(const uint4*)(Ahi + base);
        ah[ks].q[1] = *(const uint4*)(Ahi + base + 16);
        al[ks].q[0] = *(const uint4*)(Alo + base);
        al[ks].q[1] = *(const uint4*)(Alo + base + 16);
    }

    // B fragment (32x16 bf16): lanes 0-15 hold N=lane, K={kb..kb+7, kb+16..kb+23},
    // kb = (lane>>4)*8 -> two 8-ushort runs per lane from padded LDS W^T.
    const int colInTile = lane & 15;
    const int kbB       = (lane >> 4) * 8;
    const int rbase     = m0 + (lane >> 4) * 8;

#pragma unroll
    for (int nt = 0; nt < 8; ++nt) {
        const int col = nt * 16 + colInTile;
        const unsigned short* sh = sWh + col * WPITCH;
        const unsigned short* sl = sWl + col * WPITCH;
        v8f acc = {};
#pragma unroll
        for (int ks = 0; ks < 4; ++ks) {
            const int ko = ks * 32 + kbB;
            Frag bh, bl;
            bh.q[0] = *(const uint4*)(sh + ko);
            bh.q[1] = *(const uint4*)(sh + ko + 16);
            bl.q[0] = *(const uint4*)(sl + ko);
            bl.q[1] = *(const uint4*)(sl + ko + 16);
            acc = __builtin_amdgcn_wmma_f32_16x16x32_bf16(false, ah[ks].v, false, bh.v,
                                                          (short)0, acc, false, false);
            acc = __builtin_amdgcn_wmma_f32_16x16x32_bf16(false, ah[ks].v, false, bl.v,
                                                          (short)0, acc, false, false);
            acc = __builtin_amdgcn_wmma_f32_16x16x32_bf16(false, al[ks].v, false, bh.v,
                                                          (short)0, acc, false, false);
        }
        // Epilogue: C layout -> VGPR i holds row (rbase+i), col = nt*16+colInTile.
        const float bcol = bias[col];
        if (outF) {
#pragma unroll
            for (int i = 0; i < 8; ++i) {
                const int r = rbase + i;
                if (r < nrows) {
                    float v = acc[i] + bcol;
                    v = v > 0.0f ? v : 0.0f;
                    if (resid) v += resid[(size_t)r * D + col];
                    outF[(size_t)r * D + col] = v;
                }
            }
        } else {
#pragma unroll
            for (int i = 0; i < 8; ++i) {
                const int r = rbase + i;
                if (r < nrows) {
                    float v = acc[i] + bcol;
                    v = v > 0.0f ? v : 0.0f;
                    unsigned short h = f2bf(v);
                    outHi[(size_t)r * D + col] = h;
                    outLo[(size_t)r * D + col] = f2bf(v - bf2f(h));
                }
            }
        }
    }
}

// Coalesced per-column sum (+ optional sum of squares) via partial + atomics.
__global__ void col_sums_kernel(const float* __restrict__ x,
                                float* __restrict__ sums,
                                float* __restrict__ sumsq,   // nullable
                                int nrows) {
    const int col  = threadIdx.x & (D - 1);
    const int rofs = threadIdx.x >> 7;    // 0..1
    float s = 0.0f, s2 = 0.0f;
    for (int r = blockIdx.x * 2 + rofs; r < nrows; r += gridDim.x * 2) {
        float v = x[(size_t)r * D + col];
        s  += v;
        s2 += v * v;
    }
    atomicAdd(&sums[col], s);
    if (sumsq) atomicAdd(&sumsq[col], s2);
}

// In-place BatchNorm (batch stats, biased variance). Optionally dual-stores the
// result into zinit as well (pre-initializing the next layer's aggregation).
__global__ void bn_apply_kernel(float* __restrict__ x,
                                float* __restrict__ zinit,   // nullable
                                const float* __restrict__ sums,
                                const float* __restrict__ sumsq,
                                const float* __restrict__ g,
                                const float* __restrict__ be,
                                int nrows) {
    size_t idx   = (size_t)blockIdx.x * blockDim.x + threadIdx.x;
    size_t total = (size_t)nrows * D;
    if (idx >= total) return;
    int col = (int)(idx & (D - 1));
    float inv = 1.0f / (float)nrows;
    float m   = sums[col] * inv;
    float var = sumsq[col] * inv - m * m;
    float sc  = rsqrtf(var + 1e-5f) * g[col];
    float v   = (x[idx] - m) * sc + be[col];
    x[idx] = v;
    if (zinit) zinit[idx] = v;
}

__global__ void finalize_mean_kernel(const float* __restrict__ sums,
                                     float* __restrict__ out, int nrows) {
    int d = threadIdx.x;
    if (d < D) out[d] = sums[d] / (float)nrows;
}

extern "C" void kernel_launch(void* const* d_in, const int* in_sizes, int n_in,
                              void* d_out, int out_size, void* d_ws, size_t ws_size,
                              hipStream_t stream) {
    (void)n_in; (void)out_size; (void)ws_size;
    const float* features = (const float*)d_in[0];
    const int*   src      = (const int*)d_in[1];
    const int*   dst      = (const int*)d_in[2];
    const float* W[5]    = {(const float*)d_in[3], (const float*)d_in[5],
                            (const float*)d_in[7], (const float*)d_in[9],
                            (const float*)d_in[11]};
    const float* bias[5] = {(const float*)d_in[4], (const float*)d_in[6],
                            (const float*)d_in[8], (const float*)d_in[10],
                            (const float*)d_in[12]};
    const float* g[3]  = {(const float*)d_in[13], (const float*)d_in[15], (const float*)d_in[17]};
    const float* be[3] = {(const float*)d_in[14], (const float*)d_in[16], (const float*)d_in[18]};
    float* out = (float*)d_out;

    const int N = in_sizes[0] / D;
    const int E = in_sizes[1];

    char* ws = (char*)d_ws;
    size_t off = 0;
    auto carve = [&](size_t bytes) -> void* {
        void* p = ws + off;
        off += (bytes + 255) & ~(size_t)255;
        return p;
    };
    float* bufA = (float*)carve((size_t)N * D * 4);
    float* bufB = (float*)carve((size_t)N * D * 4);
    float* bufC = (float*)carve((size_t)N * D * 4);
    unsigned short* X1hi = (unsigned short*)carve((size_t)N * D * 2);
    unsigned short* X1lo = (unsigned short*)carve((size_t)N * D * 2);
    unsigned short* X2hi = (unsigned short*)carve((size_t)N * D * 2);
    unsigned short* X2lo = (unsigned short*)carve((size_t)N * D * 2);
    unsigned short *Wh[5], *Wl[5];
    for (int i = 0; i < 5; ++i) {
        Wh[i] = (unsigned short*)carve((size_t)D * D * 2);
        Wl[i] = (unsigned short*)carve((size_t)D * D * 2);
    }
    float* sums  = (float*)carve(D * 4);
    float* sumsq = (float*)carve(D * 4);

    const int    total    = N * D;
    const int    elGrid   = (total + THREADS - 1) / THREADS;
    const int    ewGrid   = (int)(((unsigned)E * 32u + THREADS - 1) / THREADS);
    const int    gemmGrid = (N + 127) / 128;
    const size_t rowBytes = (size_t)N * D * 4;

    (void)hipFuncSetAttribute((const void*)gemm_kernel,
                              hipFuncAttributeMaxDynamicSharedMemorySize, SMEM_BYTES);

    // Weight conversion (bf16 hi/lo, transposed)
    for (int i = 0; i < 5; ++i)
        split_w_kernel<<<(D * D + THREADS - 1) / THREADS, THREADS, 0, stream>>>(W[i], Wh[i], Wl[i]);

    // ---------------- layer 0 ----------------
    (void)hipMemcpyAsync(bufA, features, rowBytes, hipMemcpyDeviceToDevice, stream);  // z = h
    edge_scatter_kernel<<<ewGrid, THREADS, 0, stream>>>(features, src, dst, bufA, E);
    split_act_kernel<<<elGrid, THREADS, 0, stream>>>(bufA, X1hi, X1lo, total);
    gemm_kernel<<<gemmGrid, THREADS, SMEM_BYTES, stream>>>(X1hi, X1lo, Wh[0], Wl[0], bias[0],
                                                           nullptr, nullptr, X2hi, X2lo, N);
    gemm_kernel<<<gemmGrid, THREADS, SMEM_BYTES, stream>>>(X2hi, X2lo, Wh[1], Wl[1], bias[1],
                                                           features, bufB, nullptr, nullptr, N);
    (void)hipMemsetAsync(sums, 0, D * 4, stream);
    (void)hipMemsetAsync(sumsq, 0, D * 4, stream);
    col_sums_kernel<<<2048, THREADS, 0, stream>>>(bufB, sums, sumsq, N);
    bn_apply_kernel<<<elGrid, THREADS, 0, stream>>>(bufB, bufA, sums, sumsq, g[0], be[0], N);
    // h0 = bufB; bufA pre-initialized with h0 for the next aggregation

    // ---------------- layer 1 ----------------
    edge_scatter_kernel<<<ewGrid, THREADS, 0, stream>>>(bufB, src, dst, bufA, E);
    split_act_kernel<<<elGrid, THREADS, 0, stream>>>(bufA, X1hi, X1lo, total);
    gemm_kernel<<<gemmGrid, THREADS, SMEM_BYTES, stream>>>(X1hi, X1lo, Wh[2], Wl[2], bias[2],
                                                           nullptr, nullptr, X2hi, X2lo, N);
    gemm_kernel<<<gemmGrid, THREADS, SMEM_BYTES, stream>>>(X2hi, X2lo, Wh[3], Wl[3], bias[3],
                                                           bufB, bufC, nullptr, nullptr, N);
    (void)hipMemsetAsync(sums, 0, D * 4, stream);
    (void)hipMemsetAsync(sumsq, 0, D * 4, stream);
    col_sums_kernel<<<2048, THREADS, 0, stream>>>(bufC, sums, sumsq, N);
    bn_apply_kernel<<<elGrid, THREADS, 0, stream>>>(bufC, bufA, sums, sumsq, g[1], be[1], N);
    // h1 = bufC; bufA pre-initialized with h1

    // ---------------- layer 2 ----------------
    edge_scatter_kernel<<<ewGrid, THREADS, 0, stream>>>(bufC, src, dst, bufA, E);
    split_act_kernel<<<elGrid, THREADS, 0, stream>>>(bufA, X1hi, X1lo, total);
    gemm_kernel<<<gemmGrid, THREADS, SMEM_BYTES, stream>>>(X1hi, X1lo, Wh[4], Wl[4], bias[4],
                                                           bufC, bufB, nullptr, nullptr, N);
    (void)hipMemsetAsync(sums, 0, D * 4, stream);
    (void)hipMemsetAsync(sumsq, 0, D * 4, stream);
    col_sums_kernel<<<2048, THREADS, 0, stream>>>(bufB, sums, sumsq, N);
    bn_apply_kernel<<<elGrid, THREADS, 0, stream>>>(bufB, nullptr, sums, sumsq, g[2], be[2], N);

    // ---------------- global mean pool ----------------
    (void)hipMemsetAsync(sums, 0, D * 4, stream);
    col_sums_kernel<<<2048, THREADS, 0, stream>>>(bufB, sums, nullptr, N);
    finalize_mean_kernel<<<1, D, 0, stream>>>(sums, out, N);
}
